// KMeans_61701500175105
// MI455X (gfx1250) — matrix operand order repaired and checked by reading, not compiled, not run on hardware
//
#include <hip/hip_runtime.h>
#include <float.h>

typedef __attribute__((ext_vector_type(4)))  float    v4f;
typedef __attribute__((ext_vector_type(8)))  float    v8f;
typedef __attribute__((ext_vector_type(4)))  _Float16 v4h;
typedef __attribute__((ext_vector_type(8)))  _Float16 v8h;
typedef __attribute__((ext_vector_type(16))) _Float16 v16h;

#define BLOCK_THREADS 256
#define TILE_ROWS     128   // rows of x per block (8 waves x 16 rows)
#define KCENT         128   // number of centroids (fixed per reference)
#define DDIM          64    // feature dim (fixed per reference)
#define LDS_STRIDE    72    // halves per centroid row (64 + 8 pad): conflict-free b128 reads

__global__ __launch_bounds__(BLOCK_THREADS)
void kmeans_top2_wmma_f16x3(const float* __restrict__ x,
                            const float* __restrict__ cent,
                            float* __restrict__ out,
                            int N)
{
    __shared__ _Float16 sChi[KCENT * LDS_STRIDE];  // centroid hi halves
    __shared__ _Float16 sClo[KCENT * LDS_STRIDE];  // centroid lo halves (residual)
    __shared__ float    sCsq[KCENT];               // ||c||^2 (from hi+lo)

    const int tid  = threadIdx.x;
    const int lane = tid & 31;
    const int warp = tid >> 5;
    const int half = lane >> 4;   // 0: lanes 0-15, 1: lanes 16-31
    const int lrow = lane & 15;

    // ---- stage centroids into LDS as split-f16 (hi + lo residual) ----
    #pragma unroll
    for (int j = 0; j < (KCENT * DDIM) / (BLOCK_THREADS * 4); ++j) {
        int idx = (tid + j * BLOCK_THREADS) * 4;       // element offset, multiple of 4
        v4f f = *(const v4f*)(cent + idx);
        int row = idx >> 6;                            // idx / 64
        int col = idx & 63;
        v4h h, l;
        #pragma unroll
        for (int e = 0; e < 4; ++e) {
            _Float16 hh = (_Float16)f[e];
            h[e] = hh;
            l[e] = (_Float16)(f[e] - (float)hh);       // exact residual, then rounded
        }
        *(v4h*)(&sChi[row * LDS_STRIDE + col]) = h;
        *(v4h*)(&sClo[row * LDS_STRIDE + col]) = l;
    }
    __syncthreads();

    // ---- precompute ||c||^2 from the split representation ----
    if (tid < KCENT) {
        float s = 0.f;
        #pragma unroll 8
        for (int d = 0; d < DDIM; ++d) {
            float v = (float)sChi[tid * LDS_STRIDE + d] + (float)sClo[tid * LDS_STRIDE + d];
            s = fmaf(v, v, s);
        }
        sCsq[tid] = s;
    }
    __syncthreads();

    // ---- preload A fragments for this wave's 16 x-rows ----
    // A 16x32 f16 layout: lane L holds M=L%16; frag element e -> K = 8*half + (e<8 ? e : 8+e).
    // => each fragment = two contiguous 8-float runs of the row.
    int row  = blockIdx.x * TILE_ROWS + warp * 16 + lrow;
    int rowc = row < N ? row : N - 1;                  // clamp tail: keep EXEC all-ones
    const float* xr = x + (size_t)rowc * DDIM + 8 * half;

    float xf[32];                                      // 4 runs of 8 (run-major)
    #pragma unroll
    for (int r = 0; r < 4; ++r) {                      // run r starts at K = 16*r + 8*half
        v4f f0 = *(const v4f*)(xr + 16 * r);
        v4f f1 = *(const v4f*)(xr + 16 * r + 4);
        #pragma unroll
        for (int e = 0; e < 4; ++e) { xf[8 * r + e] = f0[e]; xf[8 * r + 4 + e] = f1[e]; }
    }

    v16h a_hi[2], a_lo[2];                             // K-chunks f=0 (K 0..31), f=1 (K 32..63)
    #pragma unroll
    for (int f = 0; f < 2; ++f) {
        #pragma unroll
        for (int e = 0; e < 16; ++e) {                 // elems 0..7 = run 2f, 8..15 = run 2f+1
            float v = xf[16 * f + e];
            _Float16 hh = (_Float16)v;
            a_hi[f][e] = hh;
            a_lo[f][e] = (_Float16)(v - (float)hh);
        }
    }

    // ---- running top-2 smallest per (row, lane) ----
    float m1[8], m2[8];
    #pragma unroll
    for (int v = 0; v < 8; ++v) { m1[v] = FLT_MAX; m2[v] = FLT_MAX; }

    // ---- sweep the 8 centroid tiles of 16 ----
    for (int tile = 0; tile < KCENT / 16; ++tile) {
        float csq = sCsq[tile * 16 + lrow];
        // B 32x16 f16 layout: lane holds N=L%16; frag element e -> K = 8*half + (e<8 ? e : 8+e)
        // (two 8-half runs). B[k][n] = cent[n][k] => runs live in centroid row n.
        const _Float16* ch = &sChi[(tile * 16 + lrow) * LDS_STRIDE + 8 * half];
        const _Float16* cl = &sClo[(tile * 16 + lrow) * LDS_STRIDE + 8 * half];

        v16h bh[2], bl[2];
        #pragma unroll
        for (int f = 0; f < 2; ++f) {                  // frag f runs at d = 32*f + 8*half (+16)
            v8h h0 = *(const v8h*)(ch + 32 * f);
            v8h h1 = *(const v8h*)(ch + 32 * f + 16);
            v8h l0 = *(const v8h*)(cl + 32 * f);
            v8h l1 = *(const v8h*)(cl + 32 * f + 16);
            #pragma unroll
            for (int e = 0; e < 8; ++e) {
                bh[f][e] = h0[e]; bh[f][8 + e] = h1[e];
                bl[f][e] = l0[e]; bl[f][8 + e] = l1[e];
            }
        }

        // f16x3: hi*hi + hi*lo + lo*hi, fp32 accumulate (lo*lo ~ 2^-22, dropped)
        v8f acc0 = {}, acc1 = {};
        acc0 = __builtin_amdgcn_wmma_f32_16x16x32_f16(false, a_hi[0], false, bh[0], (short)0, acc0, false, false);
        acc1 = __builtin_amdgcn_wmma_f32_16x16x32_f16(false, a_hi[1], false, bh[1], (short)0, acc1, false, false);
        acc0 = __builtin_amdgcn_wmma_f32_16x16x32_f16(false, a_hi[0], false, bl[0], (short)0, acc0, false, false);
        acc1 = __builtin_amdgcn_wmma_f32_16x16x32_f16(false, a_hi[1], false, bl[1], (short)0, acc1, false, false);
        acc0 = __builtin_amdgcn_wmma_f32_16x16x32_f16(false, a_lo[0], false, bh[0], (short)0, acc0, false, false);
        acc1 = __builtin_amdgcn_wmma_f32_16x16x32_f16(false, a_lo[1], false, bh[1], (short)0, acc1, false, false);

        // acc slot v holds dot(row_base + v + 8*half, tile*16 + lrow)
        #pragma unroll
        for (int v = 0; v < 8; ++v) {
            float val = fmaf(-2.f, acc0[v] + acc1[v], csq);   // ||c||^2 - 2 x.c
            if (val < m1[v]) { m2[v] = m1[v]; m1[v] = val; }
            else             { m2[v] = fminf(m2[v], val);   }
        }
    }

    // ---- butterfly top-2 merge across the 16 lanes sharing each row ----
    #pragma unroll
    for (int mask = 1; mask <= 8; mask <<= 1) {
        #pragma unroll
        for (int v = 0; v < 8; ++v) {
            float o1 = __shfl_xor(m1[v], mask, 32);
            float o2 = __shfl_xor(m2[v], mask, 32);
            float lo = fminf(m1[v], o1);
            float hi = fmaxf(m1[v], o1);
            m1[v] = lo;
            m2[v] = fminf(hi, fminf(m2[v], o2));
        }
    }

    // ---- lanes 0 and 16 of each wave write 8 rows each ----
    if (lrow == 0) {
        #pragma unroll
        for (int v = 0; v < 8; ++v) {
            int orow = blockIdx.x * TILE_ROWS + warp * 16 + half * 8 + v;
            if (orow < N) out[orow] = m2[v] - m1[v];
        }
    }
}

extern "C" void kernel_launch(void* const* d_in, const int* in_sizes, int n_in,
                              void* d_out, int out_size, void* d_ws, size_t ws_size,
                              hipStream_t stream)
{
    const float* x    = (const float*)d_in[0];   // [N, 64]
    const float* cent = (const float*)d_in[1];   // [128, 64]
    float* out        = (float*)d_out;           // [N]

    int N = in_sizes[0] / DDIM;
    int blocks = (N + TILE_ROWS - 1) / TILE_ROWS;
    hipLaunchKernelGGL(kmeans_top2_wmma_f16x3, dim3(blocks), dim3(BLOCK_THREADS), 0, stream,
                       x, cent, out, N);
}
